// ImpactModel_67396626809331
// MI455X (gfx1250) — matrix operand order repaired and checked by reading, not compiled, not run on hardware
//
#include <hip/hip_runtime.h>
#include <hip/hip_bf16.h>

// ---------------- problem constants (match reference) ----------------
#define TT   8
#define NP   16000
#define NA   16000
#define NV   200
#define DIN  256
#define HH   256
#define KA   4
#define LL   5
#define BB   (3 * NP)          // 48000 batch rows for GRU
#define SQRT2F 1.4142135623730951f

// ---------------- vector types for WMMA ----------------
typedef __attribute__((ext_vector_type(16))) __bf16 v16bf;
typedef __attribute__((ext_vector_type(8)))  __bf16 v8bf;
typedef __attribute__((ext_vector_type(8)))  float  v8f;

// ---------------- small helpers (native gfx1250 bf16 converts) --------
__device__ __forceinline__ __bf16 f2bf(float f) { return (__bf16)f; }
__device__ __forceinline__ float  bf2f(__bf16 b) { return (float)b; }
__device__ __forceinline__ float softplusf(float x) {
    return (x > 20.0f) ? x : log1pf(expf(x));
}
__device__ __forceinline__ float sigmoidf(float x) {
    return 1.0f / (1.0f + expf(-x));
}
// convert two float4 chunks into half of a v16bf fragment
__device__ __forceinline__ void cvt8(const float4 u0, const float4 u1, v16bf& a, int base) {
    a[base + 0] = f2bf(u0.x); a[base + 1] = f2bf(u0.y);
    a[base + 2] = f2bf(u0.z); a[base + 3] = f2bf(u0.w);
    a[base + 4] = f2bf(u1.x); a[base + 5] = f2bf(u1.y);
    a[base + 6] = f2bf(u1.z); a[base + 7] = f2bf(u1.w);
}
// build a v16bf fragment from two contiguous v8bf chunks
__device__ __forceinline__ v16bf frag16(const v8bf c0, const v8bf c1) {
    v16bf b;
#pragma unroll
    for (int i = 0; i < 8; ++i) { b[i] = c0[i]; b[8 + i] = c1[i]; }
    return b;
}

// =====================================================================
// Kernel 0: zero the scalar accumulators
// =====================================================================
__global__ void init_acc_kernel(float* acc) {
    if (threadIdx.x < 4) acc[threadIdx.x] = 0.0f;
}

// =====================================================================
// Kernel 1: transpose + bf16-convert weights (W^T so B-fragments are
// K-contiguous).  3x (256x256) + 2x (256x768).
// =====================================================================
__global__ void __launch_bounds__(256)
prep_weights_kernel(const float* __restrict__ Wp, const float* __restrict__ Wa,
                    const float* __restrict__ Wv, const float* __restrict__ Wx,
                    const float* __restrict__ Wh,
                    __bf16* __restrict__ WTp, __bf16* __restrict__ WTa,
                    __bf16* __restrict__ WTv, __bf16* __restrict__ WxT,
                    __bf16* __restrict__ WhT) {
    const int gid = blockIdx.x * 256 + threadIdx.x;
    const int SQ = 256 * 256;
    if (gid < 3 * SQ) {
        const int seg = gid / SQ, r = gid % SQ;
        const int nn = r / 256, kk = r % 256;
        const float* W = (seg == 0) ? Wp : (seg == 1) ? Wa : Wv;
        __bf16* O = (seg == 0) ? WTp : (seg == 1) ? WTa : WTv;
        O[nn * 256 + kk] = f2bf(W[kk * 256 + nn]);
    } else {
        const int r = gid - 3 * SQ;
        const int seg = r / (768 * 256);
        const int q = r % (768 * 256);
        const int nn = q / 256, kk = q % 256;
        const float* W = (seg == 0) ? Wx : Wh;
        __bf16* O = (seg == 0) ? WxT : WhT;
        O[nn * 256 + kk] = f2bf(W[kk * 768 + nn]);
    }
}

// =====================================================================
// Kernel 2: emb = relu(X @ W + b), X fp32 (N x 256), WT bf16 (256 x 256,
// transposed), out bf16.  32 rows/block, 8 waves, wave = (row-tile,
// 4 col-tiles), K-loop of 8 x v_wmma_f32_16x16x32_bf16.
// =====================================================================
__global__ void __launch_bounds__(256)
embed_gemm_kernel(const float* __restrict__ X, const __bf16* __restrict__ WT,
                  const float* __restrict__ bias, __bf16* __restrict__ OUT) {
    const int tid = threadIdx.x;
    const int wave = tid >> 5, lane = tid & 31, l15 = lane & 15;
    const int aoff = (lane >> 4) * 8;    // A K-offset (lane half)
    const int boff = (lane >> 4) * 8;    // B K-offset (lane half)
    const int moff = (lane >> 4) * 8;    // C/D row offset (lane half)
    const int r0 = blockIdx.x * 32 + (wave & 1) * 16;
    const int row = r0 + l15;
    const int nt0 = (wave >> 1) * 4;

    v8f acc[4] = {v8f{}, v8f{}, v8f{}, v8f{}};
    for (int kb = 0; kb < DIN; kb += 32) {
        // A fragment (16-bit A 16x32 layout), fp32 -> bf16 native cvt
        v16bf a;
        const float* ap = X + (size_t)row * DIN + kb + aoff;
        cvt8(*(const float4*)(ap),      *(const float4*)(ap + 4),  a, 0);
        cvt8(*(const float4*)(ap + 16), *(const float4*)(ap + 20), a, 8);
#pragma unroll
        for (int j = 0; j < 4; ++j) {
            const int ncol = (nt0 + j) * 16 + l15;
            const v8bf* bp = (const v8bf*)(WT + (size_t)ncol * DIN + kb + boff);
            const v16bf b = frag16(bp[0], bp[2]);
            acc[j] = __builtin_amdgcn_wmma_f32_16x16x32_bf16(
                false, a, false, b, (short)0, acc[j], false, false);
        }
    }
    // epilogue: bias + relu, store bf16
#pragma unroll
    for (int j = 0; j < 4; ++j) {
        const int ncol = (nt0 + j) * 16 + l15;
        const float bv = bias[ncol];
#pragma unroll
        for (int i = 0; i < 8; ++i) {
            float v = acc[j][i] + bv;
            v = v > 0.0f ? v : 0.0f;
            OUT[(size_t)(r0 + moff + i) * HH + ncol] = f2bf(v);
        }
    }
}

// =====================================================================
// Kernel 3: temporal smoothness loss partial sums
// =====================================================================
__global__ void __launch_bounds__(256)
ltime_kernel(const __bf16* __restrict__ emb, int n, float* __restrict__ slot) {
    const size_t per_t = (size_t)n * HH;
    const size_t gid = (size_t)blockIdx.x * 256 + threadIdx.x;
    const size_t e = gid * 8;
    float p = 0.0f;
    if (e < (size_t)(TT - 1) * per_t) {
        const size_t t = e / per_t, rem = e - t * per_t;
        const v8bf a = *(const v8bf*)(emb + t * per_t + rem);
        const v8bf c = *(const v8bf*)(emb + (t + 1) * per_t + rem);
#pragma unroll
        for (int i = 0; i < 8; ++i) {
            const float d = bf2f(c[i]) - bf2f(a[i]);
            p += d * d;
        }
    }
#pragma unroll
    for (int o = 16; o > 0; o >>= 1) p += __shfl_xor(p, o, 32);
    if ((threadIdx.x & 31) == 0) atomicAdd(slot, p);
}

// =====================================================================
// Kernel 4: neighbor gather + meta-path softmax combine
// =====================================================================
__global__ void __launch_bounds__(256)
gather_combine_kernel(const __bf16* __restrict__ emb_p,
                      const __bf16* __restrict__ emb_a,
                      const __bf16* __restrict__ emb_v,
                      const int* __restrict__ idxA, const int* __restrict__ idxV,
                      const float* __restrict__ w_meta, __bf16* __restrict__ v_all) {
    const size_t gid = (size_t)blockIdx.x * 256 + threadIdx.x;
    const int chunk = (int)(gid & 31);
    const size_t tn = gid >> 5;
    const int n = (int)(tn % NP);
    const int t = (int)(tn / NP);
    const int h0 = chunk * 8;

    const float m0 = w_meta[0], m1 = w_meta[1], m2 = w_meta[2];
    const float mx = fmaxf(m0, fmaxf(m1, m2));
    const float e0 = expf(m0 - mx), e1 = expf(m1 - mx), e2 = expf(m2 - mx);
    const float inv = 1.0f / (e0 + e1 + e2);
    const float w0 = e0 * inv, w1 = e1 * inv, w2 = e2 * inv;

    float accv[8] = {0.f, 0.f, 0.f, 0.f, 0.f, 0.f, 0.f, 0.f};
#pragma unroll
    for (int k = 0; k < KA; ++k) {
        const int ai = idxA[n * KA + k];
        const v8bf c = *(const v8bf*)(emb_a + ((size_t)t * NA + ai) * HH + h0);
#pragma unroll
        for (int i = 0; i < 8; ++i) accv[i] += bf2f(c[i]);
    }
    const int vi = idxV[n];
    const v8bf cv = *(const v8bf*)(emb_v + ((size_t)t * NV + vi) * HH + h0);
    const v8bf cp = *(const v8bf*)(emb_p + ((size_t)t * NP + n) * HH + h0);
    v8bf outv;
#pragma unroll
    for (int i = 0; i < 8; ++i) {
        const float v = w0 * 0.25f * accv[i] + w1 * bf2f(cv[i]) + w2 * bf2f(cp[i]);
        outv[i] = f2bf(v);
    }
    *(v8bf*)(v_all + ((size_t)t * NP + n) * HH + h0) = outv;
}

// =====================================================================
// Kernel 5: 5-step GRU + survival heads + prediction loss.
// 32 rows/block.  LDS: hbuf f32 (32x256), hbf bf16 (32x256),
// xbf bf16 (32x256), gh f32 (32x768) = 160 KB.
// A-fragments for both GEMM phases are pure ds_load_b128 from the bf16
// shadows -> no conversion math in the WMMA hot loops.
// =====================================================================
__global__ void __launch_bounds__(256)
gru_head_kernel(const __bf16* __restrict__ v_all, const __bf16* __restrict__ WxT,
                const __bf16* __restrict__ WhT, const float* __restrict__ b_gru,
                const float* __restrict__ W_eta, const float* __restrict__ b_eta,
                const float* __restrict__ W_mu, const float* __restrict__ b_mu,
                const float* __restrict__ W_sig, const float* __restrict__ b_sig,
                const float* __restrict__ y_true, float* __restrict__ acc) {
    extern __shared__ char smem_raw[];
    float*  hbuf = (float*)smem_raw;                       // 32 x 256 f32
    float*  gh   = hbuf + 32 * HH;                         // 32 x 768 f32
    __bf16* hbf  = (__bf16*)(gh + 32 * 768);               // 32 x 256 bf16
    __bf16* xbf  = hbf + 32 * HH;                          // 32 x 256 bf16

    const int tid = threadIdx.x;
    const int wave = tid >> 5, lane = tid & 31, l15 = lane & 15;
    const int aoff = (lane >> 4) * 8;
    const int boff = (lane >> 4) * 8;
    const int moff = (lane >> 4) * 8;
    const int b0 = blockIdx.x * 32;           // 32 | NP so yi is uniform
    const int yi = b0 / NP, n0 = b0 % NP;
    const int r0 = (wave & 1) * 16;
    const int row = r0 + l15;

    for (int i = tid; i < 32 * HH; i += 256) { hbuf[i] = 0.0f; hbf[i] = f2bf(0.0f); }
    __syncthreads();

    for (int s = 0; s < LL; ++s) {
        const int year = (5 + yi) - 1 - s;    // YEARS_TRAIN[yi] - (s+1)
        const __bf16* xbase = v_all + ((size_t)year * NP + n0) * HH;

        // ---- stage x tile into LDS (bf16, b128 copies) ----
        for (int i = tid * 8; i < 32 * HH; i += 256 * 8)
            *(v8bf*)(xbf + i) = *(const v8bf*)(xbase + i);

        // ---- gh = h @ Wh : 96 tiles / 8 waves, A from hbf (LDS) ----
        for (int j = (wave >> 1); j < 48; j += 4) {
            v8f accv = {};
            const int ncol = j * 16 + l15;
            for (int kb = 0; kb < HH; kb += 32) {
                const __bf16* hp = hbf + row * HH + kb + aoff;
                const v16bf a = frag16(*(const v8bf*)(hp), *(const v8bf*)(hp + 16));
                const v8bf* bp = (const v8bf*)(WhT + (size_t)ncol * HH + kb + boff);
                const v16bf b = frag16(bp[0], bp[2]);
                accv = __builtin_amdgcn_wmma_f32_16x16x32_bf16(
                    false, a, false, b, (short)0, accv, false, false);
            }
#pragma unroll
            for (int i = 0; i < 8; ++i)
                gh[(r0 + moff + i) * 768 + ncol] = accv[i];
        }
        __syncthreads();

        // ---- gx gate tiles (r,z,n share one A frag) + fused update ----
        for (int ct = (wave >> 1); ct < 16; ct += 4) {
            v8f ar = {}, az = {}, an = {};
            const int ncol = ct * 16 + l15;
            for (int kb = 0; kb < HH; kb += 32) {
                const __bf16* ap = xbf + row * HH + kb + aoff;
                const v16bf a = frag16(*(const v8bf*)(ap), *(const v8bf*)(ap + 16));
                const v8bf* pr = (const v8bf*)(WxT + (size_t)(ncol)       * HH + kb + boff);
                const v8bf* pz = (const v8bf*)(WxT + (size_t)(ncol + 256) * HH + kb + boff);
                const v8bf* pn = (const v8bf*)(WxT + (size_t)(ncol + 512) * HH + kb + boff);
                const v16bf br = frag16(pr[0], pr[2]);
                const v16bf bz = frag16(pz[0], pz[2]);
                const v16bf bn = frag16(pn[0], pn[2]);
                ar = __builtin_amdgcn_wmma_f32_16x16x32_bf16(false, a, false, br, (short)0, ar, false, false);
                az = __builtin_amdgcn_wmma_f32_16x16x32_bf16(false, a, false, bz, (short)0, az, false, false);
                an = __builtin_amdgcn_wmma_f32_16x16x32_bf16(false, a, false, bn, (short)0, an, false, false);
            }
            const float bvr = b_gru[ncol];
            const float bvz = b_gru[256 + ncol];
            const float bvn = b_gru[512 + ncol];
#pragma unroll
            for (int i = 0; i < 8; ++i) {
                const int rr = r0 + moff + i;
                const float hr = gh[rr * 768 + ncol];
                const float hz = gh[rr * 768 + 256 + ncol];
                const float hn = gh[rr * 768 + 512 + ncol];
                const float hp = hbuf[rr * HH + ncol];
                const float rg = sigmoidf(ar[i] + bvr + hr);
                const float zg = sigmoidf(az[i] + bvz + hz);
                const float ng = tanhf(an[i] + bvn + rg * hn);
                const float hnew = (1.0f - zg) * ng + zg * hp;
                hbuf[rr * HH + ncol] = hnew;       // f32 master copy
                hbf[rr * HH + ncol]  = f2bf(hnew); // bf16 shadow for WMMA
            }
        }
        __syncthreads();
    }

    // ---- heads + log-normal survival prediction loss ----
    float lpred = 0.0f;
#pragma unroll
    for (int q = 0; q < 4; ++q) {
        const int r = wave * 4 + q;
        float pe = 0.0f, pm = 0.0f, ps = 0.0f;
        for (int c = lane; c < HH; c += 32) {
            const float hv = hbuf[r * HH + c];
            pe += hv * W_eta[c];
            pm += hv * W_mu[c];
            ps += hv * W_sig[c];
        }
#pragma unroll
        for (int o = 16; o > 0; o >>= 1) {
            pe += __shfl_xor(pe, o, 32);
            pm += __shfl_xor(pm, o, 32);
            ps += __shfl_xor(ps, o, 32);
        }
        if (lane == 0) {
            const float eta = softplusf(pe + b_eta[0]);
            const float mu  = pm + b_mu[0];
            const float sig = softplusf(ps + b_sig[0]) + 0.001f;
            const size_t b = (size_t)b0 + r;
            float prev = 0.0f;
            for (int l = 0; l < LL; ++l) {
                const float z = (logf((float)(l + 1)) - mu) / (sig * SQRT2F);
                const float ycum = eta * 0.5f * (1.0f + erff(z));
                const float yhat = ycum - prev;
                prev = ycum;
                const float yt = y_true[b * LL + l];
                const float d = log1pf(yt + 1.0f) - log1pf(yhat);
                lpred += d * d;
            }
        }
    }
    if (lane == 0) atomicAdd(&acc[3], lpred);
}

// =====================================================================
// Kernel 6: combine partial sums -> scalar loss
// =====================================================================
__global__ void finalize_kernel(const float* __restrict__ acc, float* __restrict__ out) {
    if (threadIdx.x == 0 && blockIdx.x == 0) {
        const float lt_p = acc[0] / (float)((TT - 1) * NP);
        const float lt_a = acc[1] / (float)((TT - 1) * NA);
        const float lt_v = acc[2] / (float)((TT - 1) * NV);
        const float l_time = (lt_p + lt_a + lt_v) / 3.0f;
        const float l_pred = acc[3] / (float)(BB * LL);
        out[0] = l_pred + 0.5f * l_time;
    }
}

// =====================================================================
extern "C" void kernel_launch(void* const* d_in, const int* in_sizes, int n_in,
                              void* d_out, int out_size, void* d_ws, size_t ws_size,
                              hipStream_t stream) {
    (void)in_sizes; (void)n_in; (void)out_size; (void)ws_size;
    const float* x_paper  = (const float*)d_in[0];
    const float* x_author = (const float*)d_in[1];
    const float* x_venue  = (const float*)d_in[2];
    const int*   idxA     = (const int*)d_in[3];
    const int*   idxV     = (const int*)d_in[4];
    const float* y_true   = (const float*)d_in[5];
    const float* W_paper  = (const float*)d_in[6];
    const float* b_paper  = (const float*)d_in[7];
    const float* W_author = (const float*)d_in[8];
    const float* b_author = (const float*)d_in[9];
    const float* W_venue  = (const float*)d_in[10];
    const float* b_venue  = (const float*)d_in[11];
    const float* w_meta   = (const float*)d_in[12];
    const float* Wx       = (const float*)d_in[13];
    const float* Wh       = (const float*)d_in[14];
    const float* b_gru    = (const float*)d_in[15];
    const float* W_eta    = (const float*)d_in[16];
    const float* b_eta    = (const float*)d_in[17];
    const float* W_mu     = (const float*)d_in[18];
    const float* b_mu     = (const float*)d_in[19];
    const float* W_sigma  = (const float*)d_in[20];
    const float* b_sigma  = (const float*)d_in[21];
    float* out = (float*)d_out;

    // ---- workspace carve-up (bf16 intermediates; ~198 MB total) ----
    char* ws = (char*)d_ws;
    size_t off = 0;
    auto carve = [&](size_t bytes) -> void* {
        void* p = ws + off;
        off += (bytes + 255) & ~(size_t)255;
        return p;
    };
    __bf16* emb_p = (__bf16*)carve((size_t)TT * NP * HH * 2);
    __bf16* emb_a = (__bf16*)carve((size_t)TT * NA * HH * 2);
    __bf16* emb_v = (__bf16*)carve((size_t)TT * NV * HH * 2);
    __bf16* v_all = (__bf16*)carve((size_t)TT * NP * HH * 2);
    __bf16* WTp   = (__bf16*)carve((size_t)256 * 256 * 2);
    __bf16* WTa   = (__bf16*)carve((size_t)256 * 256 * 2);
    __bf16* WTv   = (__bf16*)carve((size_t)256 * 256 * 2);
    __bf16* WxT   = (__bf16*)carve((size_t)768 * 256 * 2);
    __bf16* WhT   = (__bf16*)carve((size_t)768 * 256 * 2);
    float*  accp  = (float*)carve(4 * sizeof(float));

    init_acc_kernel<<<1, 4, 0, stream>>>(accp);

    // 1) weight transpose + bf16 convert (589824 elems = 2304 blocks)
    prep_weights_kernel<<<2304, 256, 0, stream>>>(W_paper, W_author, W_venue, Wx, Wh,
                                                  WTp, WTa, WTv, WxT, WhT);

    // 2) embedding GEMMs (rows are exact multiples of 32)
    embed_gemm_kernel<<<(TT * NP) / 32, 256, 0, stream>>>(x_paper, WTp, b_paper, emb_p);
    embed_gemm_kernel<<<(TT * NA) / 32, 256, 0, stream>>>(x_author, WTa, b_author, emb_a);
    embed_gemm_kernel<<<(TT * NV) / 32, 256, 0, stream>>>(x_venue, WTv, b_venue, emb_v);

    // 3) temporal smoothness partial sums
    ltime_kernel<<<((TT - 1) * NP * HH / 8 + 255) / 256, 256, 0, stream>>>(emb_p, NP, accp + 0);
    ltime_kernel<<<((TT - 1) * NA * HH / 8 + 255) / 256, 256, 0, stream>>>(emb_a, NA, accp + 1);
    ltime_kernel<<<((TT - 1) * NV * HH / 8 + 255) / 256, 256, 0, stream>>>(emb_v, NV, accp + 2);

    // 4) gather + combine (T*NP*32 chunks = 16000 blocks)
    gather_combine_kernel<<<(TT * NP * (HH / 8)) / 256, 256, 0, stream>>>(
        emb_p, emb_a, emb_v, idxA, idxV, w_meta, v_all);

    // 5) GRU + heads + prediction loss  (160 KB dynamic LDS / block)
    const size_t gru_lds = (size_t)(32 * HH + 32 * 768) * sizeof(float)
                         + (size_t)(32 * HH) * 2 * 2;
    gru_head_kernel<<<BB / 32, 256, gru_lds, stream>>>(
        v_all, WxT, WhT, b_gru, W_eta, b_eta, W_mu, b_mu, W_sigma, b_sigma,
        y_true, accp);

    // 6) finalize scalar
    finalize_kernel<<<1, 1, 0, stream>>>(accp, out);
}
